// FieldCRF_27908697489726
// MI455X (gfx1250) — matrix operand order repaired and checked by reading, not compiled, not run on hardware
//
#include <hip/hip_runtime.h>
#include <hip/hip_bf16.h>

// CRF NLL on gfx1250 (MI455X, wave32).
// Forward scan as a scaled linear-domain GEMM per step using
// V_WMMA_F32_16X16X32_F16. One wave handles 16 batches; lane = batch column.
// alpha is kept in the log2 domain so p = v_exp_f32(a-m) and
// log2(sum) = v_log_f32 are single raw hardware ops (no libm guards).

typedef __attribute__((ext_vector_type(16))) _Float16 v16h;
typedef __attribute__((ext_vector_type(2)))  _Float16 v2h;
typedef __attribute__((ext_vector_type(2)))  __fp16   v2fp;
typedef __attribute__((ext_vector_type(8)))  float    v8f;

static constexpr int Bn = 1024;
static constexpr int Sn = 1024;
static constexpr int Tn = 24;

__device__ __forceinline__ float raw_exp2(float x) {
#if __has_builtin(__builtin_amdgcn_exp2f)
    return __builtin_amdgcn_exp2f(x);
#else
    return exp2f(x);
#endif
}
__device__ __forceinline__ float raw_log2(float x) {
#if __has_builtin(__builtin_amdgcn_logf)
    return __builtin_amdgcn_logf(x);
#else
    return log2f(x);
#endif
}
__device__ __forceinline__ unsigned pack_h2(float a, float b) {
#if __has_builtin(__builtin_amdgcn_cvt_pkrtz)
    union { v2fp h; unsigned u; } c;
    c.h = __builtin_amdgcn_cvt_pkrtz(a, b);
    return c.u;
#else
    union { v2h h; unsigned u; } c;
    c.h[0] = (_Float16)a; c.h[1] = (_Float16)b;
    return c.u;
#endif
}

// ---------------------------------------------------------------------------
// Kernel 1: forward scan -> log_z per batch. Grid: 64 blocks x 32 threads.
// Per-lane state (ln = lane&15, batch b = 16*block + ln), log2 domain:
//   lanes  0-15: a_lo[r]=alpha2[b][r] (states 0..7), a_hi[r]=alpha2[b][16+r]
//   lanes 16-31: a_lo[r]=alpha2[b][8+r] (states 8..15), a_hi[r]=-1e30 pad
// Matches the WMMA D layout (lane-half = M-row half) and the B layout
// (lane-half = K half): one packed shfl_xor(16) exchange per step.
// ---------------------------------------------------------------------------
__global__ void __launch_bounds__(32)
crf_scan_kernel(const float* __restrict__ logits,
                const unsigned char* __restrict__ mask,
                const float* __restrict__ trans,
                const float* __restrict__ start_t,
                const float* __restrict__ end_t,
                float* __restrict__ logz_out)
{
    constexpr float RLN2 = 1.44269504088896340736f; // 1/ln(2)
    constexpr float LN2  = 0.69314718055994530942f;

    const int  lane    = threadIdx.x;
    const int  ln      = lane & 15;
    const bool lo_half = (lane < 16);
    const int  b       = blockIdx.x * 16 + ln;

    // ---- Constant A operands: A[j][k] = exp(trans[k][j])  (E^T tiles).
    // A layout (f16 16x32): lanes 0-15 row j=ln hold K {0..15},
    // lanes 16-31 row j=ln hold K {16..31}.
    v16h A0, A1;
#pragma unroll
    for (int e = 0; e < 16; ++e) {
        const int kb = lo_half ? 0 : 16;
        const int k  = kb + e;
        float a0 = (k < Tn) ? __expf(trans[k * Tn + ln]) : 0.0f;
        float a1 = (k < Tn && ln < 8) ? __expf(trans[k * Tn + 16 + ln]) : 0.0f;
        A0[e] = (_Float16)a0;
        A1[e] = (_Float16)a1;
    }

    // ---- alpha2(t=0) = (start_transitions + logits[:,0]) / ln2
    const int off_lo = lo_half ? 0 : 8;
    const float* row0 = logits + ((size_t)b * Sn) * Tn;
    float a_lo[8], a_hi[8];
#pragma unroll
    for (int r = 0; r < 8; ++r) {
        a_lo[r] = (start_t[off_lo + r] + row0[off_lo + r]) * RLN2;
        a_hi[r] = lo_half ? (start_t[16 + r] + row0[16 + r]) * RLN2 : -1e30f;
    }
    float m = a_lo[0];
#pragma unroll
    for (int r = 1; r < 8; ++r) m = fmaxf(m, a_lo[r]);
#pragma unroll
    for (int r = 0; r < 8; ++r) m = fmaxf(m, a_hi[r]);
    m = fmaxf(m, __shfl_xor(m, 16));

    // ---- double-buffered emission row / mask pointers
    const float* row = logits + ((size_t)b * Sn + 1) * Tn;
    const unsigned char* mrow = mask + (size_t)b * Sn + 1;
    float4 c0 = *(const float4*)(row + off_lo);
    float4 c1 = *(const float4*)(row + off_lo + 4);
    float4 c2 = *(const float4*)(row + 16);
    float4 c3 = *(const float4*)(row + 20);
    unsigned char mk_cur = *mrow;

    for (int t = 1; t < Sn; ++t) {
        // preload next row (branchless tail clamp), prefetch 2 ahead
        const int adv = (t < Sn - 1) ? Tn : 0;
        const float* nrow = row + adv;
        __builtin_prefetch(row + 2 * Tn, 0, 1);
        const float4 n0 = *(const float4*)(nrow + off_lo);
        const float4 n1 = *(const float4*)(nrow + off_lo + 4);
        const float4 n2 = *(const float4*)(nrow + 16);
        const float4 n3 = *(const float4*)(nrow + 20);
        const unsigned char mk_nxt = mrow[(t < Sn - 1) ? 1 : 0];

        const float em_lo[8] = {c0.x, c0.y, c0.z, c0.w, c1.x, c1.y, c1.z, c1.w};
        const float em_hi[8] = {c2.x, c2.y, c2.z, c2.w, c3.x, c3.y, c3.z, c3.w};
        const bool  mk = (mk_cur != 0);

        // p = exp2(alpha2 - m2)  (raw v_exp_f32; pad lanes -> 0)
        float p_lo[8], p_hi[8];
#pragma unroll
        for (int r = 0; r < 8; ++r) {
            p_lo[r] = raw_exp2(a_lo[r] - m);
            p_hi[r] = raw_exp2(a_hi[r] - m);
        }

        // pack to f16 pairs, exchange 4 dwords across lane halves
        unsigned pl[4], ph[4], rcv[4];
#pragma unroll
        for (int r = 0; r < 4; ++r) {
            pl[r] = pack_h2(p_lo[2 * r], p_lo[2 * r + 1]);
            ph[r] = pack_h2(p_hi[2 * r], p_hi[2 * r + 1]);
        }
#pragma unroll
        for (int r = 0; r < 4; ++r) {
            unsigned snd = lo_half ? ph[r] : pl[r];
            rcv[r] = (unsigned)__shfl_xor((int)snd, 16);
        }

        // B operand (32x16 f16): lane-half = K block of 16; K>=24 zero pad
        union { v16h h; unsigned u[8]; } Bu;
#pragma unroll
        for (int r = 0; r < 4; ++r) {
            Bu.u[r]     = lo_half ? pl[r]  : rcv[r];
            Bu.u[4 + r] = lo_half ? rcv[r] : 0u;
        }

        // D = E^T x P : two M-tiles of output states
        v8f cz = {};
        v8f d1 = __builtin_amdgcn_wmma_f32_16x16x32_f16(
            false, A0, false, Bu.h, (short)0, cz, false, false);
        v8f d2 = __builtin_amdgcn_wmma_f32_16x16x32_f16(
            false, A1, false, Bu.h, (short)0, cz, false, false);

        // alpha2' = em/ln2 + m2 + log2(sum); fold next max into update
        float mn = -1e30f;
#pragma unroll
        for (int r = 0; r < 8; ++r) {
            float na = fmaf(em_lo[r], RLN2, m + raw_log2(d1[r]));
            a_lo[r] = mk ? na : a_lo[r];
            mn = fmaxf(mn, a_lo[r]);
        }
#pragma unroll
        for (int r = 0; r < 8; ++r) {
            float na = fmaf(em_hi[r], RLN2, m + raw_log2(d2[r]));
            a_hi[r] = (mk && lo_half) ? na : a_hi[r];
            mn = fmaxf(mn, a_hi[r]);
        }
        m = fmaxf(mn, __shfl_xor(mn, 16));

        // rotate buffers
        c0 = n0; c1 = n1; c2 = n2; c3 = n3;
        mk_cur = mk_nxt;
        row = nrow;
        mrow += (t < Sn - 1) ? 1 : 0;
    }

    // ---- log_z = ln2 * log2sumexp2(alpha2 + end/ln2)
    float v_lo[8], v_hi[8];
#pragma unroll
    for (int r = 0; r < 8; ++r) {
        v_lo[r] = fmaf(end_t[off_lo + r], RLN2, a_lo[r]);
        v_hi[r] = lo_half ? fmaf(end_t[16 + r], RLN2, a_hi[r]) : -1e30f;
    }
    float m2 = v_lo[0];
#pragma unroll
    for (int r = 1; r < 8; ++r) m2 = fmaxf(m2, v_lo[r]);
#pragma unroll
    for (int r = 0; r < 8; ++r) m2 = fmaxf(m2, v_hi[r]);
    m2 = fmaxf(m2, __shfl_xor(m2, 16));
    float s = 0.0f;
#pragma unroll
    for (int r = 0; r < 8; ++r)
        s += raw_exp2(v_lo[r] - m2) + raw_exp2(v_hi[r] - m2);
    s += __shfl_xor(s, 16);
    if (lo_half) logz_out[b] = (m2 + raw_log2(s)) * LN2;
}

// ---------------------------------------------------------------------------
// Kernel 2: supervised score path. One block per batch, threads stride S.
// ---------------------------------------------------------------------------
__global__ void __launch_bounds__(128)
crf_score_kernel(const float* __restrict__ logits,
                 const int* __restrict__ labels,
                 const unsigned char* __restrict__ mask,
                 const float* __restrict__ trans,
                 const float* __restrict__ start_t,
                 const float* __restrict__ end_t,
                 float* __restrict__ score_out)
{
    const int b   = blockIdx.x;
    const int tid = threadIdx.x;
    float s = 0.0f;
    int cnt = 0;
    for (int t = tid; t < Sn; t += 128) {
        const int lbl = labels[(size_t)b * Sn + t];
        const int mk  = mask[(size_t)b * Sn + t];
        cnt += mk;
        if (t == 0) {
            s += start_t[lbl] + logits[((size_t)b * Sn) * Tn + lbl];
        } else if (mk) {
            const int prev = labels[(size_t)b * Sn + t - 1];
            s += trans[prev * Tn + lbl] + logits[((size_t)b * Sn + t) * Tn + lbl];
        }
    }
    __shared__ float sbuf[128];
    __shared__ int   cbuf[128];
    sbuf[tid] = s;
    cbuf[tid] = cnt;
    __syncthreads();
#pragma unroll
    for (int st = 64; st > 0; st >>= 1) {
        if (tid < st) {
            sbuf[tid] += sbuf[tid + st];
            cbuf[tid] += cbuf[tid + st];
        }
        __syncthreads();
    }
    if (tid == 0) {
        const int last_idx = cbuf[0] - 1;
        const int last_tag = labels[(size_t)b * Sn + last_idx];
        score_out[b] = sbuf[0] + end_t[last_tag];
    }
}

// ---------------------------------------------------------------------------
// Kernel 3: out = -mean(score - log_z)
// ---------------------------------------------------------------------------
__global__ void __launch_bounds__(256)
crf_reduce_kernel(const float* __restrict__ score,
                  const float* __restrict__ logz,
                  float* __restrict__ out, int n)
{
    __shared__ float buf[256];
    const int tid = threadIdx.x;
    float s = 0.0f;
    for (int i = tid; i < n; i += 256) s += score[i] - logz[i];
    buf[tid] = s;
    __syncthreads();
#pragma unroll
    for (int st = 128; st > 0; st >>= 1) {
        if (tid < st) buf[tid] += buf[tid + st];
        __syncthreads();
    }
    if (tid == 0) out[0] = -buf[0] / (float)n;
}

extern "C" void kernel_launch(void* const* d_in, const int* in_sizes, int n_in,
                              void* d_out, int out_size, void* d_ws, size_t ws_size,
                              hipStream_t stream)
{
    (void)in_sizes; (void)n_in; (void)out_size; (void)ws_size;
    const float*         logits  = (const float*)d_in[0];
    const int*           labels  = (const int*)d_in[1];
    const unsigned char* mask    = (const unsigned char*)d_in[2]; // jnp bool_ = 1B
    const float*         trans   = (const float*)d_in[3];
    const float*         start_t = (const float*)d_in[4];
    const float*         end_t   = (const float*)d_in[5];

    float* logz  = (float*)d_ws;   // [Bn]
    float* score = logz + Bn;      // [Bn]

    crf_scan_kernel<<<Bn / 16, 32, 0, stream>>>(logits, mask, trans, start_t,
                                                end_t, logz);
    crf_score_kernel<<<Bn, 128, 0, stream>>>(logits, labels, mask, trans,
                                             start_t, end_t, score);
    crf_reduce_kernel<<<1, 256, 0, stream>>>(score, logz, (float*)d_out, Bn);
}
